// DPINet_82867099009817
// MI455X (gfx1250) — matrix operand order repaired and checked by reading, not compiled, not run on hardware
//
#include <hip/hip_runtime.h>
#include <hip/hip_bf16.h>
#include <stdint.h>

typedef __bf16 bf16t;
typedef __attribute__((ext_vector_type(16))) __bf16 v16bf;
typedef __attribute__((ext_vector_type(8)))  __bf16 v8bf;
typedef __attribute__((ext_vector_type(8)))  float  v8f;

#define NN    50000
#define NEDGE 400000
#define EST   (NEDGE / 2)
#define HID   128

__constant__ float d_std[6] = {1.f, 1.f, 1.f, 0.1f, 0.1f, 0.1f};

// ---------------------------------------------------------------- utilities
__global__ void k_zero32(uint32_t* p, size_t n) {
  size_t i = (size_t)blockIdx.x * blockDim.x + threadIdx.x;
  size_t st = (size_t)gridDim.x * blockDim.x;
  for (; i < n; i += st) p[i] = 0u;
}

__global__ void k_f2b(const float* __restrict__ s, bf16t* __restrict__ d, size_t n) {
  size_t i = (size_t)blockIdx.x * blockDim.x + threadIdx.x;
  size_t st = (size_t)gridDim.x * blockDim.x;
  for (; i < n; i += st) d[i] = (bf16t)s[i];
}

// Pack W[K x 128] (f32) into B-fragment layout: [kc][nc][lane][e] bf16, zero-pad K.
__global__ void k_pack(const float* __restrict__ W, bf16t* __restrict__ Wp, int K, int total) {
  int t = blockIdx.x * blockDim.x + threadIdx.x;
  if (t >= total) return;
  int e    = t & 15;
  int lane = (t >> 4) & 31;
  int nc   = (t >> 9) & 7;
  int kc   = t >> 12;
  int n = nc * 16 + (lane & 15);
  int k = kc * 32 + ((lane >> 4) ? 8 : 0) + (e & 7) + ((e >> 3) << 4);
  float v = (k < K) ? W[(size_t)k * HID + n] : 0.f;
  Wp[t] = (bf16t)v;
}

// ---------------------------------------------------------------- pre-pass
__global__ __launch_bounds__(192) void k_inst_mean(const float* __restrict__ nodes,
                                                   const int* __restrict__ instIdx,
                                                   float* __restrict__ instMean,
                                                   float* __restrict__ counts) {
  __shared__ float sh[192];
  int inst = blockIdx.x;
  int lo = instIdx[inst], hi = instIdx[inst + 1];
  int t = threadIdx.x;
  int d = t % 6, slice = t / 6;
  float s = 0.f;
  for (int r = lo + slice; r < hi; r += 32) s += nodes[(size_t)r * 6 + d] / d_std[d];
  sh[t] = s;
  __syncthreads();
  if (slice == 0) {
    float a = 0.f;
    for (int k = 0; k < 32; ++k) a += sh[k * 6 + d];
    instMean[inst * 6 + d] = a / (float)(hi - lo);
    if (d == 0) counts[inst] = (float)(hi - lo);
  }
}

// featA[n][0:6]=nodes_n, [6:15]=node_attrs, [15:21]=offsets, [21:32]=0 (bf16)
__global__ void k_feat(const float* __restrict__ nodes, const float* __restrict__ nattr,
                       const int* __restrict__ instIdx, const float* __restrict__ instMean,
                       float* __restrict__ nodesN, bf16t* __restrict__ featA, int nInst) {
  int n = blockIdx.x * blockDim.x + threadIdx.x;
  if (n >= NN) return;
  int g = 0;
  while (g + 1 < nInst && instIdx[g + 1] <= n) ++g;
  bf16t* f = featA + (size_t)n * 32;
  for (int d = 0; d < 6; ++d) {
    float v = nodes[(size_t)n * 6 + d] / d_std[d];
    nodesN[(size_t)n * 6 + d] = v;
    f[d] = (bf16t)v;
    f[15 + d] = (bf16t)(v - instMean[g * 6 + d]);
  }
  for (int d = 0; d < 9; ++d) f[6 + d] = (bf16t)nattr[(size_t)n * 9 + d];
  for (int j = 21; j < 32; ++j) f[j] = (bf16t)0.f;
}

// rel_enc input row: [feat_r(21) | feat_s(21) | rel_attr(4) | pad..64]
__global__ void k_edge_in(const int* __restrict__ rels, const float* __restrict__ relAttr,
                          const bf16t* __restrict__ featA, bf16t* __restrict__ Xe, int base) {
  int e = blockIdx.x * blockDim.x + threadIdx.x;
  if (e >= EST) return;
  int ge = base + e;
  int r = rels[2 * ge], s = rels[2 * ge + 1];
  bf16t* o = Xe + (size_t)e * 64;
  const bf16t* fr = featA + (size_t)r * 32;
  const bf16t* fs = featA + (size_t)s * 32;
  for (int j = 0; j < 21; ++j) o[j] = fr[j];
  for (int j = 0; j < 21; ++j) o[21 + j] = fs[j];
  for (int j = 0; j < 4; ++j)  o[42 + j] = (bf16t)relAttr[(size_t)ge * 4 + j];
  for (int j = 46; j < 64; ++j) o[j] = (bf16t)0.f;
}

__global__ void k_mask(const int* __restrict__ rels, int* __restrict__ mask, int base) {
  int e = blockIdx.x * blockDim.x + threadIdx.x;
  if (e >= EST) return;
  mask[rels[2 * (base + e)]] = 1;
}

// ---------------------------------------------------------------- WMMA GEMM
// EPI: 0=bf16 store, 1=bf16 store + ReLU, 2=atomic scatter-add f32 by ridx, 3=masked effects update
// GUARD=false: wave-uniform full tile, no per-element row bounds checks (no exec churn).
template<int EPI, bool GUARD>
__device__ __forceinline__ void gemm_epilogue(
    const v8f (&acc)[8], int rowBase, int mOff, int m, int rows,
    const float* __restrict__ bias, const int* __restrict__ rels, int edgeBase,
    bf16t* __restrict__ Yb, float* __restrict__ Yf,
    float* __restrict__ effF, bf16t* __restrict__ effB, const int* __restrict__ mask) {
  int rIdx[8];
  if constexpr (EPI == 2) {
#pragma unroll
    for (int r = 0; r < 8; ++r) {
      const int row = rowBase + mOff + r;
      rIdx[r] = (!GUARD || row < rows) ? rels[2 * (edgeBase + row)] : 0;
    }
  }
#pragma unroll
  for (int nc = 0; nc < 8; ++nc) {
    const int col = nc * 16 + m;
    const float bcol = bias[col];
#pragma unroll
    for (int r = 0; r < 8; ++r) {
      const int row = rowBase + mOff + r;
      if constexpr (GUARD) { if (row >= rows) continue; }
      float v = acc[nc][r] + bcol;
      if constexpr (EPI == 1) { v = v > 0.f ? v : 0.f; Yb[(size_t)row * HID + col] = (bf16t)v; }
      else if constexpr (EPI == 0) { Yb[(size_t)row * HID + col] = (bf16t)v; }
      else if constexpr (EPI == 2) { atomicAdd(Yf + (size_t)rIdx[r] * HID + col, v); }
      else {
        if (mask[row]) {
          const size_t o = (size_t)row * HID + col;
          const float nv = effF[o] + v;
          effF[o] = nv;
          effB[o] = (bf16t)nv;
        }
      }
    }
  }
}

// Y[rows x 128] = epilogue(X[rows x K] @ W + bias)
// AMODE: 0=dense X, 1=cat3 gathered by edge (er|es|rel_enc), 2=cat3 identity (ne|eff|agg)
template<int KCH, int AMODE, int EPI>
__global__ __launch_bounds__(256) void k_gemm(
    const bf16t* __restrict__ A0, const bf16t* __restrict__ A1, const bf16t* __restrict__ A2,
    const int* __restrict__ rels, int edgeBase,
    const bf16t* __restrict__ Wp, const float* __restrict__ bias, int rows,
    bf16t* __restrict__ Yb, float* __restrict__ Yf,
    float* __restrict__ effF, bf16t* __restrict__ effB, const int* __restrict__ mask) {
  const int lane = threadIdx.x & 31;
  const int wave = threadIdx.x >> 5;
  const int rowBase = blockIdx.x * 128 + wave * 16;
  const int m = lane & 15;
  const int half = lane >> 4;
  int arow = rowBase + m;
  if (arow > rows - 1) arow = rows - 1;

  const bf16t* pr0 = nullptr; const bf16t* pr1 = nullptr; const bf16t* pr2 = nullptr;
  if constexpr (AMODE == 0) {
    pr0 = A0 + (size_t)arow * (KCH * 32);
  } else if constexpr (AMODE == 1) {
    const int ge = edgeBase + arow;
    pr0 = A0 + (size_t)rels[2 * ge] * HID;
    pr1 = A1 + (size_t)rels[2 * ge + 1] * HID;
    pr2 = A2 + (size_t)arow * HID;
  } else {
    pr0 = A0 + (size_t)arow * HID;
    pr1 = A1 + (size_t)arow * HID;
    pr2 = A2 + (size_t)arow * HID;
  }

  v8f acc[8] = {};

#pragma unroll
  for (int kc = 0; kc < KCH; ++kc) {
    const bf16t* src;
    int ko;
    if constexpr (AMODE == 0) { src = pr0; ko = kc * 32; }
    else { const int seg = kc >> 2; src = (seg == 0) ? pr0 : ((seg == 1) ? pr1 : pr2); ko = (kc & 3) * 32; }
    // 16-bit A 16x32 fragment: lanes<16 hold K {ko..ko+7, ko+16..ko+23}; lanes>=16 shift by 8
    const bf16t* p = src + ko + half * 8;
    v8bf lo = *(const v8bf*)(p);
    v8bf hi = *(const v8bf*)(p + 16);
    v16bf a = __builtin_shufflevector(lo, hi, 0,1,2,3,4,5,6,7,8,9,10,11,12,13,14,15);
#pragma unroll
    for (int nc = 0; nc < 8; ++nc) {
      v16bf b = *(const v16bf*)(Wp + ((((size_t)kc * 8 + nc) * 32 + lane) << 4));
      acc[nc] = __builtin_amdgcn_wmma_f32_16x16x32_bf16(false, a, false, b, (short)0, acc[nc], false, false);
    }
  }

  const int mOff = half * 8;
  if (rowBase + 16 <= rows) {
    gemm_epilogue<EPI, false>(acc, rowBase, mOff, m, rows, bias, rels, edgeBase, Yb, Yf, effF, effB, mask);
  } else {
    gemm_epilogue<EPI, true>(acc, rowBase, mOff, m, rows, bias, rels, edgeBase, Yb, Yf, effF, effB, mask);
  }
}

// ---------------------------------------------------------------- tail
__global__ __launch_bounds__(128) void k_eff_mean(const float* __restrict__ effF,
                                                  const int* __restrict__ instIdx,
                                                  const float* __restrict__ counts,
                                                  float* __restrict__ effMean) {
  int inst = blockIdx.x, col = threadIdx.x;
  int lo = instIdx[inst], hi = instIdx[inst + 1];
  float s = 0.f;
  for (int r = lo; r < hi; ++r) s += effF[(size_t)r * HID + col];
  effMean[inst * HID + col] = s / counts[inst];
}

__global__ __launch_bounds__(128) void k_head(
    const float* __restrict__ effMean, const float* __restrict__ instMean,
    const float* __restrict__ W0, const float* __restrict__ b0,
    const float* __restrict__ W1, const float* __restrict__ b1,
    const float* __restrict__ W2, const float* __restrict__ b2,
    const float* __restrict__ dtPtr, float* __restrict__ rtc, int nInst) {
  __shared__ float X[16 * HID];
  __shared__ float Y[16 * HID];
  __shared__ float Pm[16 * 9];
  const int t = threadIdx.x;
  for (int i = t; i < nInst * HID; i += 128) X[i] = effMean[i];
  __syncthreads();
  for (int i = 0; i < nInst; ++i) {
    float s = b0[t];
    for (int k = 0; k < HID; ++k) s += X[i * HID + k] * W0[(size_t)k * HID + t];
    Y[i * HID + t] = s > 0.f ? s : 0.f;
  }
  __syncthreads();
  for (int i = t; i < nInst * HID; i += 128) X[i] = Y[i];
  __syncthreads();
  for (int i = 0; i < nInst; ++i) {
    float s = b1[t];
    for (int k = 0; k < HID; ++k) s += X[i * HID + k] * W1[(size_t)k * HID + t];
    Y[i * HID + t] = s > 0.f ? s : 0.f;
  }
  __syncthreads();
  for (int i = t; i < nInst * HID; i += 128) X[i] = Y[i];
  __syncthreads();
  if (t < 9) {
    const float dt = *dtPtr;
    for (int i = 0; i < nInst; ++i) {
      float s = b2[t];
      for (int k = 0; k < HID; ++k) s += X[i * HID + k] * W2[(size_t)k * 9 + t];
      Pm[i * 9 + t] = s * dt;
    }
  }
  __syncthreads();
  if (t < nInst) {
    const float* p = Pm + t * 9;
    float a1x = p[3], a1y = p[4], a1z = p[5], a2x = p[6], a2y = p[7], a2z = p[8];
    float n1 = sqrtf(a1x * a1x + a1y * a1y + a1z * a1z);
    float b1x = a1x / n1, b1y = a1y / n1, b1z = a1z / n1;
    float dd = b1x * a2x + b1y * a2y + b1z * a2z;
    float c2x = a2x - dd * b1x, c2y = a2y - dd * b1y, c2z = a2z - dd * b1z;
    float n2 = sqrtf(c2x * c2x + c2y * c2y + c2z * c2z);
    float b2x = c2x / n2, b2y = c2y / n2, b2z = c2z / n2;
    float b3x = b1y * b2z - b1z * b2y, b3y = b1z * b2x - b1x * b2z, b3z = b1x * b2y - b1y * b2x;
    float* o = rtc + t * 15;
    o[0] = b1x; o[1] = b1y; o[2] = b1z;
    o[3] = b2x; o[4] = b2y; o[5] = b2z;
    o[6] = b3x; o[7] = b3y; o[8] = b3z;
    o[9]  = p[0] * d_std[0]; o[10] = p[1] * d_std[1]; o[11] = p[2] * d_std[2];
    o[12] = instMean[t * 6 + 0] * d_std[0];
    o[13] = instMean[t * 6 + 1] * d_std[1];
    o[14] = instMean[t * 6 + 2] * d_std[2];
  }
}

__global__ void k_final(const float* __restrict__ nodesN, const int* __restrict__ instIdx,
                        const float* __restrict__ rtc, const float* __restrict__ dtPtr,
                        float* __restrict__ out, int nInst) {
  int n = blockIdx.x * blockDim.x + threadIdx.x;
  if (n >= NN) return;
  int g = 0;
  while (g + 1 < nInst && instIdx[g + 1] <= n) ++g;
  const float* o = rtc + g * 15;
  const float dt = *dtPtr;
  float p0x = nodesN[(size_t)n * 6 + 0] * d_std[0];
  float p0y = nodesN[(size_t)n * 6 + 1] * d_std[1];
  float p0z = nodesN[(size_t)n * 6 + 2] * d_std[2];
  float vx = p0x - o[12], vy = p0y - o[13], vz = p0z - o[14];
  float p1x = vx * o[0] + vy * o[3] + vz * o[6] + o[9]  + o[12];
  float p1y = vx * o[1] + vy * o[4] + vz * o[7] + o[10] + o[13];
  float p1z = vx * o[2] + vy * o[5] + vz * o[8] + o[11] + o[14];
  out[(size_t)n * 3 + 0] = (p1x - p0x) / dt;
  out[(size_t)n * 3 + 1] = (p1y - p0y) / dt;
  out[(size_t)n * 3 + 2] = (p1z - p0z) / dt;
}

// ---------------------------------------------------------------- host
extern "C" void kernel_launch(void* const* d_in, const int* in_sizes, int n_in,
                              void* d_out, int out_size, void* d_ws, size_t ws_size,
                              hipStream_t stream) {
  const float* nodes   = (const float*)d_in[0];
  const float* nattr   = (const float*)d_in[1];
  const int*   rels    = (const int*)d_in[2];
  const float* relAttr = (const float*)d_in[3];
  const int*   instIdx = (const int*)d_in[5];
  const float* dtPtr   = (const float*)d_in[7];
  const int nInst = in_sizes[5] - 1;

  // ---- resolve param leaf order into insertion-order table prm[46]
  // insertion idx: node_enc s*6+l (0..11), rel_enc 12+s*6+l, rel_prop 24+s*4+l,
  //                node_prop 32+s*4+l, node_out 40+l
  const float* prm[46];
  if (n_in >= 54) {
    bool insertion = (in_sizes[20] == 46 * 128);
    if (insertion) {
      for (int i = 0; i < 46; ++i) prm[i] = (const float*)d_in[8 + i];
    } else { // sorted keys: node_enc, node_out, node_prop, rel_enc, rel_prop
      int map[46]; int p = 0;
      for (int i = 0;  i < 12; ++i) map[p++] = i;
      for (int i = 40; i < 46; ++i) map[p++] = i;
      for (int i = 32; i < 40; ++i) map[p++] = i;
      for (int i = 12; i < 24; ++i) map[p++] = i;
      for (int i = 24; i < 32; ++i) map[p++] = i;
      for (int i = 0; i < 46; ++i) prm[map[i]] = (const float*)d_in[8 + i];
    }
  } else { // single concatenated blob, insertion order
    static const int sz[46] = {2688,128,16384,128,16384,128, 2688,128,16384,128,16384,128,
                               5888,128,16384,128,16384,128, 5888,128,16384,128,16384,128,
                               49152,128,16384,128, 49152,128,16384,128,
                               49152,128,16384,128, 49152,128,16384,128,
                               16384,128,16384,128,1152,9};
    const float* base = (const float*)d_in[8];
    size_t off = 0;
    for (int i = 0; i < 46; ++i) { prm[i] = base + off; off += sz[i]; }
  }

  // ---- workspace carve-up
  char* w = (char*)d_ws;
  auto alloc = [&](size_t bytes) -> char* { char* p = w; w += (bytes + 255) & ~(size_t)255; return p; };
  float* nodesN   = (float*)alloc((size_t)NN * 6 * 4);
  float* instMean = (float*)alloc((size_t)nInst * 6 * 4);
  float* counts   = (float*)alloc((size_t)nInst * 4);
  bf16t* featA    = (bf16t*)alloc((size_t)NN * 32 * 2);
  bf16t* Xe       = (bf16t*)alloc((size_t)EST * 64 * 2);
  bf16t* ne       = (bf16t*)alloc((size_t)NN * HID * 2);
  bf16t* re       = (bf16t*)alloc((size_t)EST * HID * 2);
  bf16t* he       = (bf16t*)alloc((size_t)EST * HID * 2);
  bf16t* hn       = (bf16t*)alloc((size_t)NN * HID * 2);
  float* effF     = (float*)alloc((size_t)NN * HID * 4);
  bf16t* effB     = (bf16t*)alloc((size_t)NN * HID * 2);
  float* aggF     = (float*)alloc((size_t)NN * HID * 4);
  bf16t* aggB     = (bf16t*)alloc((size_t)NN * HID * 2);
  int*   mask     = (int*)alloc((size_t)NN * 4);
  float* effMean  = (float*)alloc((size_t)nInst * HID * 4);
  float* rtc      = (float*)alloc((size_t)nInst * 15 * 4);
  bf16t* wNE[2][3]; bf16t* wRE[2][3]; bf16t* wRP[2][2]; bf16t* wNP[2][2];
  for (int s = 0; s < 2; ++s) {
    wNE[s][0] = (bf16t*)alloc(1  * 4096 * 2);
    wNE[s][1] = (bf16t*)alloc(4  * 4096 * 2);
    wNE[s][2] = (bf16t*)alloc(4  * 4096 * 2);
    wRE[s][0] = (bf16t*)alloc(2  * 4096 * 2);
    wRE[s][1] = (bf16t*)alloc(4  * 4096 * 2);
    wRE[s][2] = (bf16t*)alloc(4  * 4096 * 2);
    wRP[s][0] = (bf16t*)alloc(12 * 4096 * 2);
    wRP[s][1] = (bf16t*)alloc(4  * 4096 * 2);
    wNP[s][0] = (bf16t*)alloc(12 * 4096 * 2);
    wNP[s][1] = (bf16t*)alloc(4  * 4096 * 2);
  }
  (void)ws_size; (void)out_size;

  auto zero32 = [&](void* p, size_t words) {
    int blocks = (int)((words + 255) / 256); if (blocks > 4096) blocks = 4096;
    k_zero32<<<blocks, 256, 0, stream>>>((uint32_t*)p, words);
  };
  auto pack = [&](const float* W, bf16t* dst, int K, int Kch) {
    int total = Kch * 4096;
    k_pack<<<(total + 255) / 256, 256, 0, stream>>>(W, dst, K, total);
  };

  // ---- pack all weights (biases stay f32, read directly)
  for (int s = 0; s < 2; ++s) {
    pack(prm[s * 6 + 0],      wNE[s][0], 21, 1);
    pack(prm[s * 6 + 2],      wNE[s][1], 128, 4);
    pack(prm[s * 6 + 4],      wNE[s][2], 128, 4);
    pack(prm[12 + s * 6 + 0], wRE[s][0], 46, 2);
    pack(prm[12 + s * 6 + 2], wRE[s][1], 128, 4);
    pack(prm[12 + s * 6 + 4], wRE[s][2], 128, 4);
    pack(prm[24 + s * 4 + 0], wRP[s][0], 384, 12);
    pack(prm[24 + s * 4 + 2], wRP[s][1], 128, 4);
    pack(prm[32 + s * 4 + 0], wNP[s][0], 384, 12);
    pack(prm[32 + s * 4 + 2], wNP[s][1], 128, 4);
  }

  // ---- pre-pass
  zero32(effF, (size_t)NN * HID);
  zero32(effB, (size_t)NN * HID / 2);
  k_inst_mean<<<nInst, 192, 0, stream>>>(nodes, instIdx, instMean, counts);
  k_feat<<<(NN + 255) / 256, 256, 0, stream>>>(nodes, nattr, instIdx, instMean, nodesN, featA, nInst);

  const int gN = (NN + 127) / 128;
  const int gE = (EST + 127) / 128;

  for (int s = 0; s < 2; ++s) {
    const int base = s * EST;
    const float* neB0 = prm[s * 6 + 1]; const float* neB1 = prm[s * 6 + 3]; const float* neB2 = prm[s * 6 + 5];
    const float* reB0 = prm[12 + s * 6 + 1]; const float* reB1 = prm[12 + s * 6 + 3]; const float* reB2 = prm[12 + s * 6 + 5];
    const float* rpB0 = prm[24 + s * 4 + 1]; const float* rpB1 = prm[24 + s * 4 + 3];
    const float* npB0 = prm[32 + s * 4 + 1]; const float* npB1 = prm[32 + s * 4 + 3];

    zero32(mask, NN);
    k_mask<<<(EST + 255) / 256, 256, 0, stream>>>(rels, mask, base);

    // node_enc (per node): featA -> hn -> hn -> ne
    k_gemm<1, 0, 1><<<gN, 256, 0, stream>>>(featA, nullptr, nullptr, nullptr, 0, wNE[s][0], neB0, NN, hn, nullptr, nullptr, nullptr, nullptr);
    k_gemm<4, 0, 1><<<gN, 256, 0, stream>>>(hn,    nullptr, nullptr, nullptr, 0, wNE[s][1], neB1, NN, hn, nullptr, nullptr, nullptr, nullptr);
    k_gemm<4, 0, 0><<<gN, 256, 0, stream>>>(hn,    nullptr, nullptr, nullptr, 0, wNE[s][2], neB2, NN, ne, nullptr, nullptr, nullptr, nullptr);

    // rel_enc (per edge): Xe -> he -> he -> re
    k_edge_in<<<(EST + 255) / 256, 256, 0, stream>>>(rels, relAttr, featA, Xe, base);
    k_gemm<2, 0, 1><<<gE, 256, 0, stream>>>(Xe, nullptr, nullptr, nullptr, 0, wRE[s][0], reB0, EST, he, nullptr, nullptr, nullptr, nullptr);
    k_gemm<4, 0, 1><<<gE, 256, 0, stream>>>(he, nullptr, nullptr, nullptr, 0, wRE[s][1], reB1, EST, he, nullptr, nullptr, nullptr, nullptr);
    k_gemm<4, 0, 0><<<gE, 256, 0, stream>>>(he, nullptr, nullptr, nullptr, 0, wRE[s][2], reB2, EST, re, nullptr, nullptr, nullptr, nullptr);

    for (int step = 0; step < 2; ++step) {
      zero32(aggF, (size_t)NN * HID);
      // rel_prop L0: gather [er|es|rel_enc] -> he (ReLU)
      k_gemm<12, 1, 1><<<gE, 256, 0, stream>>>(effB, effB, re, rels, base, wRP[s][0], rpB0, EST, he, nullptr, nullptr, nullptr, nullptr);
      // rel_prop L1: he -> scatter-add into aggF by receiver (fused segment_sum)
      k_gemm<4, 0, 2><<<gE, 256, 0, stream>>>(he, nullptr, nullptr, rels, base, wRP[s][1], rpB1, EST, nullptr, aggF, nullptr, nullptr, nullptr);
      k_f2b<<<4096, 256, 0, stream>>>(aggF, aggB, (size_t)NN * HID);
      // node_prop L0: [ne|eff|agg] -> hn (ReLU)
      k_gemm<12, 2, 1><<<gN, 256, 0, stream>>>(ne, effB, aggB, nullptr, 0, wNP[s][0], npB0, NN, hn, nullptr, nullptr, nullptr, nullptr);
      // node_prop L1: hn -> masked effects += delta (f32 master + bf16 mirror)
      k_gemm<4, 0, 3><<<gN, 256, 0, stream>>>(hn, nullptr, nullptr, nullptr, 0, wNP[s][1], npB1, NN, nullptr, nullptr, effF, effB, mask);
    }
  }

  // ---- tail: instance mean -> output head -> rigid transform
  k_eff_mean<<<nInst, 128, 0, stream>>>(effF, instIdx, counts, effMean);
  k_head<<<1, 128, 0, stream>>>(effMean, instMean,
                                prm[40], prm[41], prm[42], prm[43], prm[44], prm[45],
                                dtPtr, rtc, nInst);
  k_final<<<(NN + 255) / 256, 256, 0, stream>>>(nodesN, instIdx, rtc, dtPtr, (float*)d_out, nInst);
}
